// AdditiveAttention_5815385719145
// MI455X (gfx1250) — matrix operand order repaired and checked by reading, not compiled, not run on hardware
//
#include <hip/hip_runtime.h>
#include <hip/hip_bf16.h>
#include <math.h>

// Problem sizes (compile-time constants from the reference).
#define BATCH 64
#define SEQ   2048
#define DIM   512
#define MROWS (BATCH * SEQ)          // 131072 rows of the flattened GEMM
#define ROWS_PER_BLOCK 128           // 8 waves x 16 rows
#define LDS_STRIDE 520               // 512 + 8 bf16 pad -> b128 LDS reads conflict-free

typedef __attribute__((ext_vector_type(16))) __bf16 bfx16;
typedef __attribute__((ext_vector_type(8)))  __bf16 bfx8;
typedef __attribute__((ext_vector_type(4)))  __bf16 bfx4;
typedef __attribute__((ext_vector_type(8)))  float  fx8;

__device__ __forceinline__ bfx16 cat8(bfx8 lo, bfx8 hi) {
    bfx16 r;
#pragma unroll
    for (int i = 0; i < 8; ++i) { r[i] = lo[i]; r[i + 8] = hi[i]; }
    return r;
}

// gfx1250 has a hardware V_TANH_F32 (TRANS op, co-executes with VALU/WMMA).
__device__ __forceinline__ float fast_tanh(float v) {
#if __has_builtin(__builtin_amdgcn_tanhf)
    return __builtin_amdgcn_tanhf(v);
#else
    return tanhf(v);
#endif
}

// ---------------------------------------------------------------------------
// Kernel 0: convert W1 [512x512] f32 -> bf16 in workspace (L2-resident, 512KB)
// ---------------------------------------------------------------------------
__global__ void cvt_w1_kernel(const float* __restrict__ w1, __bf16* __restrict__ w1bf) {
    int idx = (blockIdx.x * 256 + threadIdx.x) * 4;   // 256 blocks cover 262144 elems
    float4 v = *(const float4*)(w1 + idx);
    bfx4 o;
    o[0] = (__bf16)v.x; o[1] = (__bf16)v.y; o[2] = (__bf16)v.z; o[3] = (__bf16)v.w;
    *(bfx4*)(w1bf + idx) = o;
}

// ---------------------------------------------------------------------------
// Kernel 1: scores[m] = sum_e tanh( (x @ W1^T)[m,e] + b1[e] ) * w2[e] + b2
// WMMA bf16 GEMM, h never materialized in memory.
// ---------------------------------------------------------------------------
__global__ void score_kernel(const float* __restrict__ x,
                             const __bf16* __restrict__ w1bf,
                             const float* __restrict__ b1,
                             const float* __restrict__ w2,
                             const float* __restrict__ b2p,
                             float* __restrict__ scores) {
    extern __shared__ __align__(16) char dyn_smem[];
    __bf16* xs = (__bf16*)dyn_smem;                 // 128 x LDS_STRIDE bf16 = 130KB

    const int tid = threadIdx.x;
    const int row_base = blockIdx.x * ROWS_PER_BLOCK;

    // -------- Phase 1: stage 128 rows of x into LDS as bf16 (row-major, padded)
    const float4* xg = (const float4*)(x + (size_t)row_base * DIM);
#pragma unroll 4
    for (int it = 0; it < 64; ++it) {               // 16384 float4 / 256 threads
        int idx  = it * 256 + tid;
        int fidx = idx * 4;
        int r = fidx >> 9;                           // /512
        int c = fidx & (DIM - 1);
        float4 v = xg[idx];
        bfx4 o;
        o[0] = (__bf16)v.x; o[1] = (__bf16)v.y; o[2] = (__bf16)v.z; o[3] = (__bf16)v.w;
        *(bfx4*)(xs + r * LDS_STRIDE + c) = o;
    }
    __syncthreads();

    // -------- Phase 2: each wave owns a 16-row A tile, sweeps 32 e-tiles
    const int wave = tid >> 5;
    const int lane = tid & 31;
    const int m    = lane & 15;     // row (A) / column (B,C,D) within tile
    const int hi   = lane >> 4;     // half-wave select per ISA VGPR layouts

    // A-layout (16x32 bf16): lane<16 holds K = {k0..k0+7, k0+16..k0+23},
    //                        lane>=16 holds K = {k0+8..k0+15, k0+24..k0+31}
    const __bf16* arow = xs + (size_t)(wave * 16 + m) * LDS_STRIDE + hi * 8;

    float sacc[8];
#pragma unroll
    for (int i = 0; i < 8; ++i) sacc[i] = 0.0f;

    for (int et = 0; et < 32; ++et) {
        const int e0 = et * 16;
        const float b1v = b1[e0 + m];
        const float w2v = w2[e0 + m];
        // B-layout (32x16 bf16): lane n holds column n = W1 row (e0+n),
        // same K order as A: lanes 0-15 -> K={k0..k0+7, k0+16..k0+23}, lanes 16-31 -> K={k0+8..k0+15, k0+24..k0+31}
        const __bf16* brow = w1bf + (size_t)(e0 + m) * DIM + hi * 8;

        fx8 c8 = {0.f, 0.f, 0.f, 0.f, 0.f, 0.f, 0.f, 0.f};
#pragma unroll 4
        for (int k0 = 0; k0 < DIM; k0 += 32) {
            bfx8 alo = *(const bfx8*)(arow + k0);
            bfx8 ahi = *(const bfx8*)(arow + k0 + 16);
            bfx8 blo = *(const bfx8*)(brow + k0);
            bfx8 bhi = *(const bfx8*)(brow + k0 + 16);
            bfx16 a = cat8(alo, ahi);
            bfx16 b = cat8(blo, bhi);
            c8 = __builtin_amdgcn_wmma_f32_16x16x32_bf16(
                     /*neg_a=*/false, a, /*neg_b=*/false, b,
                     /*c_mod=*/(short)0, c8, /*reuse_a=*/false, /*reuse_b=*/false);
        }
        // C/D layout: VGPR i, lanes 0-15 -> row i, lanes 16-31 -> row i+8; col = lane%16
#pragma unroll
        for (int i = 0; i < 8; ++i)
            sacc[i] += fast_tanh(c8[i] + b1v) * w2v;
    }

    // Reduce the 16 column-lanes of each half-wave (rows 0-7 in lanes 0-15,
    // rows 8-15 in lanes 16-31).
#pragma unroll
    for (int i = 0; i < 8; ++i) {
        float v = sacc[i];
        v += __shfl_xor(v, 1, 16);
        v += __shfl_xor(v, 2, 16);
        v += __shfl_xor(v, 4, 16);
        v += __shfl_xor(v, 8, 16);
        sacc[i] = v;
    }
    if (m == 0) {
        const float b2v = b2p[0];
#pragma unroll
        for (int i = 0; i < 8; ++i) {
            int row = row_base + wave * 16 + hi * 8 + i;
            scores[row] = sacc[i] + b2v;
        }
    }
}

// ---------------------------------------------------------------------------
// Kernel 2: in-place softmax over S=2048 per batch row (scores -> weights)
// ---------------------------------------------------------------------------
__global__ void softmax_kernel(float* __restrict__ wout) {
    __shared__ float red[256];
    const int b = blockIdx.x;
    const int tid = threadIdx.x;
    float* row = wout + (size_t)b * SEQ;

    float v[8];
    float mx = -3.402823466e38f;
#pragma unroll
    for (int i = 0; i < 8; ++i) { v[i] = row[tid + i * 256]; mx = fmaxf(mx, v[i]); }
    red[tid] = mx;
    __syncthreads();
    for (int off = 128; off > 0; off >>= 1) {
        if (tid < off) red[tid] = fmaxf(red[tid], red[tid + off]);
        __syncthreads();
    }
    mx = red[0];
    __syncthreads();

    float s = 0.0f;
#pragma unroll
    for (int i = 0; i < 8; ++i) { v[i] = expf(v[i] - mx); s += v[i]; }
    red[tid] = s;
    __syncthreads();
    for (int off = 128; off > 0; off >>= 1) {
        if (tid < off) red[tid] += red[tid + off];
        __syncthreads();
    }
    const float inv = 1.0f / red[0];
#pragma unroll
    for (int i = 0; i < 8; ++i) row[tid + i * 256] = v[i] * inv;
}

// ---------------------------------------------------------------------------
// Kernel 3: ctx[b,d] = sum_s x[b,s,d] * w[b,s]   (coalesced, no atomics)
// blockIdx.x = b*4 + colchunk; 256 threads: 128 columns x 2 s-slices
// ---------------------------------------------------------------------------
__global__ void ctx_kernel(const float* __restrict__ x,
                           const float* __restrict__ wsm,
                           float* __restrict__ ctx) {
    __shared__ float red[256];
    const int b   = blockIdx.x >> 2;
    const int c0  = (blockIdx.x & 3) * 128;
    const int tid = threadIdx.x;
    const int col = c0 + (tid & 127);
    const int ss  = tid >> 7;                        // 0 or 1
    const float* xb = x + (size_t)b * SEQ * DIM;
    const float* wb = wsm + (size_t)b * SEQ;

    float acc = 0.0f;
    const int s_beg = ss * 1024, s_end = s_beg + 1024;
    for (int s = s_beg; s < s_end; ++s)
        acc += xb[(size_t)s * DIM + col] * wb[s];

    red[tid] = acc;
    __syncthreads();
    if (tid < 128)
        ctx[(size_t)b * DIM + col] = red[tid] + red[tid + 128];
}

// ---------------------------------------------------------------------------
extern "C" void kernel_launch(void* const* d_in, const int* in_sizes, int n_in,
                              void* d_out, int out_size, void* d_ws, size_t ws_size,
                              hipStream_t stream) {
    const float* x  = (const float*)d_in[0];   // [B,S,D]
    const float* W1 = (const float*)d_in[1];   // [D,D]
    const float* b1 = (const float*)d_in[2];   // [D]
    const float* w2 = (const float*)d_in[3];   // [D]
    const float* b2 = (const float*)d_in[4];   // scalar

    float* out  = (float*)d_out;
    float* ctx  = out;                          // [B,D]   = 32768 floats
    float* wout = out + (size_t)BATCH * DIM;    // [B,S,1] = 131072 floats

    __bf16* w1bf = (__bf16*)d_ws;               // 512KB bf16 copy of W1

    // 0) W1 -> bf16 (L2-resident across all GEMM blocks)
    cvt_w1_kernel<<<256, 256, 0, stream>>>(W1, w1bf);

    // 1) WMMA score GEMM: 130KB dynamic LDS per block (gfx1250 WGP has 320KB)
    const size_t lds_bytes = (size_t)ROWS_PER_BLOCK * LDS_STRIDE * sizeof(__bf16);
    (void)hipFuncSetAttribute((const void*)score_kernel,
                              hipFuncAttributeMaxDynamicSharedMemorySize,
                              (int)lds_bytes);
    score_kernel<<<MROWS / ROWS_PER_BLOCK, 256, lds_bytes, stream>>>(
        x, w1bf, b1, w2, b2, wout);

    // 2) softmax over S, in place in the output's w region
    softmax_kernel<<<BATCH, 256, 0, stream>>>(wout);

    // 3) weighted sum -> ctx
    ctx_kernel<<<BATCH * 4, 256, 0, stream>>>(x, wout, ctx);
}
